// GcnEncoderCell_89687507075331
// MI455X (gfx1250) — compile-verified, matrix operand order checked by reading, not run on hardware
//
#include <hip/hip_runtime.h>
#include <cstdint>
#include <cstddef>

// B=64, N=325, T=12, D=64, HEADS=8, HOPS=3, RANK=10
#define NN 325
#define NP 352            // N padded to multiple of 32 (11 K-steps)
#define TT 12
#define DD 64
#define BB 64
#define KSTEPS_A 11       // NP/32
#define MT 22             // NP/16 (real M tiles)
#define MTP 24            // M tiles padded to multiple of 4 (A-frags zero-padded)
#define CW 768            // T*D: flattened (t,d) column width of hop state
#define HSZ ((size_t)BB*NP*CW)   // elements per hop buffer

typedef __attribute__((ext_vector_type(16))) __bf16 v16bf;
typedef __attribute__((ext_vector_type(8)))  float  v8f;

__device__ __forceinline__ __bf16 f2bf(float f){
  unsigned u = __builtin_bit_cast(unsigned, f);
  u += 0x7fffu + ((u >> 16) & 1u);          // round-to-nearest-even
  unsigned short s = (unsigned short)(u >> 16);
  return __builtin_bit_cast(__bf16, s);
}
__device__ __forceinline__ float bf2f(__bf16 h){
  unsigned short s = __builtin_bit_cast(unsigned short, h);
  unsigned u = ((unsigned)s) << 16;
  return __builtin_bit_cast(float, u);
}

__device__ __forceinline__ v8f wmma_bf16(v16bf a, v16bf b, v8f c){
  // D = A(16x32 bf16) x B(32x16 bf16) + C(16x16 f32)
  return __builtin_amdgcn_wmma_f32_16x16x32_bf16(false, a, false, b, (short)0, c, false, false);
}

// A-fragment (16x32 bf16) from row-major bf16 [16][stride]:
// lane l holds row l&15; K interleave: {koff..koff+7, 16+koff..16+koff+7}, koff=(l>=16)*8
__device__ __forceinline__ v16bf lda_frag(const __bf16* base, int stride){
  int lane = threadIdx.x & 31;
  const __bf16* p = base + (lane & 15) * stride + ((lane >> 4) << 3);
  v16bf a;
#pragma unroll
  for (int j = 0; j < 8; j++){ a[j] = p[j]; a[8+j] = p[16+j]; }
  return a;
}
// Pre-swizzled fragment: lane reads 16 contiguous bf16 (32B) -> 2x global_load_b128
__device__ __forceinline__ v16bf ld_frag_sw(const __bf16* chunk){
  const __bf16* p = chunk + (threadIdx.x & 31) * 16;
  v16bf r;
#pragma unroll
  for (int j = 0; j < 16; j++) r[j] = p[j];
  return r;
}

// ---------------- prep: fold weights + swizzle to fragment layout ----------------
__device__ void fold_kc(const float* A, const float* Bw, int K, float* WS){
  // WS[k][c] = sum_r A[k][r] * Bw[r][c]   (A: [K,64], Bw: [64,64])
  for (int i = threadIdx.x; i < K*64; i += blockDim.x){
    int k = i >> 6, c = i & 63;
    float a = 0.f;
    for (int r = 0; r < 64; r++) a += A[k*64 + r] * Bw[r*64 + c];
    WS[i] = a;
  }
}
__device__ void swz_store(const float* W, int K, __bf16* dst){
  // dst layout: [K/32 ksteps][4 col-tiles][32 lanes][16 bf16]
  int nk = K >> 5;
  for (int it = threadIdx.x; it < nk*128; it += blockDim.x){
    int ks = it >> 7; int rem = it & 127; int ct = rem >> 5; int lane = rem & 31;
    int col = ct*16 + (lane & 15); int kh = (lane >> 4) << 3;
    __bf16* p = dst + (size_t)it * 16;
#pragma unroll
    for (int j = 0; j < 8; j++){
      p[j]   = f2bf(W[(ks*32 + kh + j)      * 64 + col]);
      p[8+j] = f2bf(W[(ks*32 + 16 + kh + j) * 64 + col]);
    }
  }
}
__device__ void fold_bias(const float* bvec, const float* Bw, const float* b2, float* outb){
  for (int c = threadIdx.x; c < 64; c += blockDim.x){
    float a = b2[c];
    for (int r = 0; r < 64; r++) a += bvec[r] * Bw[r*64 + c];
    outb[c] = a;
  }
}
__global__ void prep_kernel(const float* f1_w, const float* f1_b,
                            const float* f2_w, const float* f2_b,
                            const float* f3_w, const float* f3_b,
                            const float* q_w, const float* q_b,
                            const float* k_w, const float* k_b,
                            const float* v_w, const float* v_b,
                            const float* o_w, const float* spaceF_w,
                            __bf16* wq, __bf16* wk, __bf16* wv, __bf16* wo, __bf16* wg,
                            float* bias){
  __shared__ float WS[128*64];
  fold_kc(f1_w, q_w, 128, WS); __syncthreads();
  swz_store(WS, 128, wq); fold_bias(f1_b, q_w, q_b, bias); __syncthreads();
  fold_kc(f2_w, k_w, 128, WS); __syncthreads();
  swz_store(WS, 128, wk); fold_bias(f2_b, k_w, k_b, bias + 64); __syncthreads();
  fold_kc(f3_w, v_w, 64, WS); __syncthreads();
  swz_store(WS, 64, wv); fold_bias(f3_b, v_w, v_b, bias + 128);
  swz_store(o_w, 64, wo);
  swz_store(spaceF_w, 128, wg);
}

// ---------------- adjacency: A = softmax(relu(m1@m2)), emitted as A-fragments ----------------
// grid = MTP*2 blocks of 8 rows; rows >= NN (incl. the 2 pad M-tiles) emit zeros.
__global__ void adj_kernel(const float* m1, const float* m2, __bf16* afrag){
  __shared__ float S[8][NP];
  __shared__ float smax[8], sinv[8];
  int r0 = blockIdx.x * 8;
  for (int i = threadIdx.x; i < 8*NP; i += blockDim.x){
    int r = i / NP, n = i % NP; int m = r0 + r;
    float s = 0.f;
    if (m < NN && n < NN){
      float a = 0.f;
#pragma unroll
      for (int j = 0; j < 10; j++) a += m1[m*10 + j] * m2[j*NN + n];
      s = fmaxf(a, 0.f);
    }
    S[r][n] = s;
  }
  __syncthreads();
  if (threadIdx.x < 8){
    int r = threadIdx.x; int m = r0 + r;
    float mx = -1e30f, sum = 0.f;
    if (m < NN){
      for (int n = 0; n < NN; n++) mx = fmaxf(mx, S[r][n]);
      for (int n = 0; n < NN; n++) sum += __expf(S[r][n] - mx);
    }
    smax[r] = mx; sinv[r] = (m < NN) ? 1.f / sum : 0.f;
  }
  __syncthreads();
  for (int i = threadIdx.x; i < 8*NP; i += blockDim.x){
    int r = i / NP, n = i % NP; int m = r0 + r;
    float p = (m < NN && n < NN) ? __expf(S[r][n] - smax[r]) * sinv[r] : 0.f;
    S[r][n] = p;
  }
  __syncthreads();
  int mtile = r0 >> 4; int rlo = (blockIdx.x & 1) * 8;
  for (int it = threadIdx.x; it < KSTEPS_A*16; it += blockDim.x){
    int kk = it >> 4; int qq = it & 15; int hi = qq >> 3; int rsub = qq & 7;
    int lane = hi*16 + rlo + rsub;
    int kh = hi * 8;
    __bf16* p = afrag + ((size_t)(mtile*KSTEPS_A + kk) * 32 + lane) * 16;
#pragma unroll
    for (int j = 0; j < 8; j++){
      p[j]   = f2bf(S[rsub][kk*32 + kh + j]);
      p[8+j] = f2bf(S[rsub][kk*32 + 16 + kh + j]);
    }
  }
}

// ---------------- zero the K-pad rows of H0 ----------------
__global__ void padzero_kernel(__bf16* H0){
  size_t total = (size_t)BB * (NP - NN) * CW;
  for (size_t i = (size_t)blockIdx.x * blockDim.x + threadIdx.x; i < total;
       i += (size_t)gridDim.x * blockDim.x){
    size_t b = i / ((NP - NN) * CW); size_t rem = i % ((NP - NN) * CW);
    size_t r = rem / CW; size_t c = rem % CW;
    H0[(b*NP + NN + r) * CW + c] = f2bf(0.f);
  }
}

// ---------------- fused q/k/v/g projections + causal attention + output proj ----------------
__global__ void __launch_bounds__(128) attn_kernel(
    const float* x, const float* hidden, const float* tXin,
    const __bf16* wq, const __bf16* wk, const __bf16* wv,
    const __bf16* wo, const __bf16* wg,
    const float* bias, const float* spaceF_b, const float* o_b,
    __bf16* H0, float* out){
  __shared__ __bf16 xh[16][128];    // [x | hidden]
  __shared__ __bf16 hx[16][128];    // [hidden | tXin]
  __shared__ float  qs[16][DD], kqs[16][DD], vs[16][DD];
  __shared__ __bf16 ao[16][DD];
  int bn = blockIdx.x;
  int b = bn / NN, n = bn % NN;
  size_t base = (size_t)bn * CW;
  for (int i = threadIdx.x; i < 16*64; i += blockDim.x){
    int t = i >> 6, d = i & 63;
    float xv = 0.f, hv = 0.f, tv = 0.f;
    if (t < TT){ xv = x[base + t*64 + d]; hv = hidden[base + t*64 + d]; tv = tXin[base + t*64 + d]; }
    xh[t][d] = f2bf(xv); xh[t][64 + d] = f2bf(hv);
    hx[t][d] = f2bf(hv); hx[t][64 + d] = f2bf(tv);
  }
  __syncthreads();
  int lane = threadIdx.x & 31, wid = threadIdx.x >> 5;
  int col = wid*16 + (lane & 15);
  int rbase = (lane >> 4) << 3;
  v8f qc, kc, vc, gc;
  {
    float bq = bias[col], bk = bias[64 + col], bv = bias[128 + col], bg = spaceF_b[col];
#pragma unroll
    for (int r = 0; r < 8; r++){ qc[r] = bq; kc[r] = bk; vc[r] = bv; gc[r] = bg; }
  }
#pragma unroll
  for (int ksx = 0; ksx < 4; ksx++){
    v16bf ah = lda_frag(&hx[0][0] + ksx*32, 128);
    qc = wmma_bf16(ah, ld_frag_sw(wq + (size_t)(ksx*4 + wid) * 512), qc);
    kc = wmma_bf16(ah, ld_frag_sw(wk + (size_t)(ksx*4 + wid) * 512), kc);
    v16bf ax = lda_frag(&xh[0][0] + ksx*32, 128);
    gc = wmma_bf16(ax, ld_frag_sw(wg + (size_t)(ksx*4 + wid) * 512), gc);
  }
#pragma unroll
  for (int ksx = 0; ksx < 2; ksx++){
    v16bf ah = lda_frag(&hx[0][0] + ksx*32, 128);    // hidden = cols 0..63 of hx
    vc = wmma_bf16(ah, ld_frag_sw(wv + (size_t)(ksx*4 + wid) * 512), vc);
  }
  __bf16* h0p = H0 + ((size_t)b*NP + n) * CW;
#pragma unroll
  for (int r = 0; r < 8; r++){
    int row = rbase + r;
    qs[row][col] = qc[r]; kqs[row][col] = kc[r]; vs[row][col] = vc[r];
    if (row < TT) h0p[row*64 + col] = f2bf(gc[r]);     // g -> H0 (bf16 hop state)
  }
  for (int i = threadIdx.x; i < 4*64; i += blockDim.x) ao[12 + (i >> 6)][i & 63] = f2bf(0.f);
  __syncthreads();
  if (threadIdx.x < 96){                                // 8 heads x 12 query rows
    int h = threadIdx.x / 12, t = threadIdx.x % 12;
    const float scale = 0.35355339059327373f;           // 1/sqrt(8)
    float sc[TT]; float mx = -1e30f;
    for (int s = 0; s <= t; s++){
      float a = 0.f;
#pragma unroll
      for (int d = 0; d < 8; d++) a += qs[t][h*8 + d] * kqs[s][h*8 + d];
      a *= scale;
      sc[s] = a; mx = fmaxf(mx, a);
    }
    float sum = 0.f;
    for (int s = 0; s <= t; s++){ sc[s] = __expf(sc[s] - mx); sum += sc[s]; }
    float inv = 1.f / sum;
#pragma unroll
    for (int d = 0; d < 8; d++){
      float o = 0.f;
      for (int s = 0; s <= t; s++) o += sc[s] * vs[s][h*8 + d];
      ao[t][h*8 + d] = f2bf(o * inv);
    }
  }
  __syncthreads();
  v8f oc;
  { float bo = o_b[col];
#pragma unroll
    for (int r = 0; r < 8; r++) oc[r] = bo; }
#pragma unroll
  for (int ksx = 0; ksx < 2; ksx++){
    v16bf aa = lda_frag(&ao[0][0] + ksx*32, 64);
    oc = wmma_bf16(aa, ld_frag_sw(wo + (size_t)(ksx*4 + wid) * 512), oc);
  }
  float* op = out + base;
#pragma unroll
  for (int r = 0; r < 8; r++){
    int row = rbase + r;
    if (row < TT) op[row*64 + col] = oc[r];             // attention branch -> d_out
  }
}

// ---------------- hop: H_dst[b] = A @ H_src[b]  ([352,352]@[352,768] bf16 WMMA) ----------------
// Block computes a 64x64 tile: 4 waves x (own M-tile) x 4 column tiles.
// K panels (32x64 bf16, 4KB) are copied global->LDS with GLOBAL_LOAD_ASYNC_TO_LDS_B128
// (ASYNCcnt-tracked DMA, no VGPR round trip), then consumed as B-fragments.
#define PSTR 72   // row-major panel stride in elems (144B rows -> every 16B chunk aligned)
__global__ void __launch_bounds__(128) hop_kernel(const __bf16* __restrict__ afrag,
                                                  const __bf16* __restrict__ src,
                                                  __bf16* __restrict__ dst){
  __shared__ __bf16 panel[32 * PSTR];
  int cg = blockIdx.x;          // 12 column groups of 64
  int mt0 = blockIdx.y * 4;     // 6 groups of 4 M-tiles (2 pad tiles are zero A)
  int b = blockIdx.z;
  int cbase = cg * 64;
  const __bf16* Hs = src + (size_t)b * NP * CW;
  int lane = threadIdx.x & 31, wid = threadIdx.x >> 5;
  int mtile = mt0 + wid;
  v8f acc0 = {}, acc1 = {}, acc2 = {}, acc3 = {};
  // two 16B chunks per thread cover the 4KB panel (256 chunks)
  int k0 = threadIdx.x >> 3,        c0 = (threadIdx.x & 7) * 8;
  int k1 = (threadIdx.x + 128) >> 3, c1 = (threadIdx.x & 7) * 8;
  unsigned l0 = (unsigned)(uintptr_t)&panel[k0*PSTR + c0];
  unsigned l1 = (unsigned)(uintptr_t)&panel[k1*PSTR + c1];
  for (int kk = 0; kk < KSTEPS_A; kk++){
    __syncthreads();
    {
      const __bf16* g0 = Hs + (size_t)(kk*32 + k0) * CW + cbase + c0;
      const __bf16* g1 = Hs + (size_t)(kk*32 + k1) * CW + cbase + c1;
      asm volatile("global_load_async_to_lds_b128 %0, %1, off"
                   :: "v"(l0), "v"(g0) : "memory");
      asm volatile("global_load_async_to_lds_b128 %0, %1, off"
                   :: "v"(l1), "v"(g1) : "memory");
      asm volatile("s_wait_asynccnt 0x0" ::: "memory");
    }
    __syncthreads();
    v16bf a = ld_frag_sw(afrag + (size_t)(mtile*KSTEPS_A + kk) * 512);
    int koff = (lane >> 4) << 3;
    int csub = lane & 15;
#pragma unroll
    for (int ct = 0; ct < 4; ct++){
      int col = ct*16 + csub;
      v16bf bb;
#pragma unroll
      for (int j = 0; j < 8; j++){
        bb[j]   = panel[(koff + j) * PSTR + col];
        bb[8+j] = panel[(16 + koff + j) * PSTR + col];
      }
      if (ct == 0)      acc0 = wmma_bf16(a, bb, acc0);
      else if (ct == 1) acc1 = wmma_bf16(a, bb, acc1);
      else if (ct == 2) acc2 = wmma_bf16(a, bb, acc2);
      else              acc3 = wmma_bf16(a, bb, acc3);
    }
  }
  if (mtile < MT){
    int rb = mtile*16 + ((lane >> 4) << 3);
    int cs = cbase + (lane & 15);
    __bf16* Hd = dst + (size_t)b * NP * CW;
#pragma unroll
    for (int r = 0; r < 8; r++){
      size_t rowoff = (size_t)(rb + r) * CW;
      Hd[rowoff + cs     ] = f2bf(acc0[r]);
      Hd[rowoff + cs + 16] = f2bf(acc1[r]);
      Hd[rowoff + cs + 32] = f2bf(acc2[r]);
      Hd[rowoff + cs + 48] = f2bf(acc3[r]);
    }
  }
}

// ---------------- combine: gcn mix (K=48) + add attention + sigmoid*tanh ----------------
__global__ void __launch_bounds__(256) combine_kernel(const __bf16* __restrict__ Hall,
                                                      const float* gcn_w, const float* gcn_b,
                                                      float* out){
  __shared__ __bf16 h4[4 * CW];
  __shared__ float gw[48 * 12];
  __shared__ float gb[12];
  int bn = blockIdx.x; int b = bn / NN, n = bn % NN;
  size_t hb = ((size_t)b * NP + n) * CW;
  for (int i = threadIdx.x; i < 4*CW; i += 256){
    int hop = i / CW, c = i % CW;
    h4[i] = Hall[(size_t)hop * HSZ + hb + c];
  }
  for (int i = threadIdx.x; i < 48*12; i += 256) gw[i] = gcn_w[i];
  if (threadIdx.x < 12) gb[threadIdx.x] = gcn_b[threadIdx.x];
  __syncthreads();
  size_t ob = (size_t)bn * CW;
  for (int o = threadIdx.x; o < CW; o += 256){
    int to = o >> 6, d = o & 63;
    float acc = gb[to];
#pragma unroll
    for (int hop = 0; hop < 4; hop++)
#pragma unroll
      for (int ti = 0; ti < 12; ti++)
        acc += bf2f(h4[hop*CW + ti*64 + d]) * gw[(hop*12 + ti)*12 + to];
    float z = acc + out[ob + o];
    float sg = 1.f / (1.f + __expf(-z));
    out[ob + o] = sg * tanhf(z);
  }
}

extern "C" void kernel_launch(void* const* d_in, const int* in_sizes, int n_in,
                              void* d_out, int out_size, void* d_ws, size_t ws_size,
                              hipStream_t stream){
  (void)in_sizes; (void)n_in; (void)out_size; (void)ws_size;
  const float* x        = (const float*)d_in[0];
  const float* hidden   = (const float*)d_in[1];
  const float* tXin     = (const float*)d_in[2];
  const float* m1       = (const float*)d_in[3];
  const float* m2       = (const float*)d_in[4];
  const float* spaceF_w = (const float*)d_in[5];
  const float* spaceF_b = (const float*)d_in[6];
  const float* gcn_w    = (const float*)d_in[7];
  const float* gcn_b    = (const float*)d_in[8];
  const float* f1_w = (const float*)d_in[9];
  const float* f1_b = (const float*)d_in[10];
  const float* f2_w = (const float*)d_in[11];
  const float* f2_b = (const float*)d_in[12];
  const float* f3_w = (const float*)d_in[13];
  const float* f3_b = (const float*)d_in[14];
  const float* q_w  = (const float*)d_in[15];
  const float* q_b  = (const float*)d_in[16];
  const float* k_w  = (const float*)d_in[17];
  const float* k_b  = (const float*)d_in[18];
  const float* v_w  = (const float*)d_in[19];
  const float* v_b  = (const float*)d_in[20];
  const float* o_w  = (const float*)d_in[21];
  const float* o_b  = (const float*)d_in[22];
  float* out = (float*)d_out;

  uint8_t* ws = (uint8_t*)d_ws;
  size_t off = 0;
  auto alloc = [&](size_t bytes) -> void* {
    void* p = ws + off; off += (bytes + 255) & ~(size_t)255; return p;
  };
  __bf16* wq    = (__bf16*)alloc(16384);
  __bf16* wk    = (__bf16*)alloc(16384);
  __bf16* wv    = (__bf16*)alloc(8192);
  __bf16* wo    = (__bf16*)alloc(8192);
  __bf16* wg    = (__bf16*)alloc(16384);
  float*  bias  = (float*)alloc(3*64*sizeof(float));
  __bf16* afrag = (__bf16*)alloc((size_t)MTP * KSTEPS_A * 512 * 2);
  __bf16* Hall  = (__bf16*)alloc((size_t)4 * HSZ * 2);
  __bf16* H0 = Hall;
  __bf16* H1 = Hall + HSZ;
  __bf16* H2 = Hall + 2*HSZ;
  __bf16* H3 = Hall + 3*HSZ;

  prep_kernel<<<1, 256, 0, stream>>>(f1_w, f1_b, f2_w, f2_b, f3_w, f3_b,
                                     q_w, q_b, k_w, k_b, v_w, v_b,
                                     o_w, spaceF_w, wq, wk, wv, wo, wg, bias);
  adj_kernel<<<MTP*2, 256, 0, stream>>>(m1, m2, afrag);   // 48 blocks (incl. zero pad tiles)
  padzero_kernel<<<512, 256, 0, stream>>>(H0);
  attn_kernel<<<BB*NN, 128, 0, stream>>>(x, hidden, tXin, wq, wk, wv, wo, wg,
                                         bias, spaceF_b, o_b, H0, out);
  dim3 hgrid(CW/64, MTP/4, BB);
  hop_kernel<<<hgrid, 128, 0, stream>>>(afrag, H0, H1);
  hop_kernel<<<hgrid, 128, 0, stream>>>(afrag, H1, H2);
  hop_kernel<<<hgrid, 128, 0, stream>>>(afrag, H2, H3);
  combine_kernel<<<BB*NN, 256, 0, stream>>>(Hall, gcn_w, gcn_b, out);
}
